// transformer_137438953771
// MI455X (gfx1250) — matrix operand order repaired and checked by reading, not compiled, not run on hardware
//
#include <hip/hip_runtime.h>
#include <hip/hip_bf16.h>

// ---------------------------------------------------------------------------
// CDNA5 (gfx1250) types / helpers: wave32 WMMA 16x16x32 bf16 -> f32
// ---------------------------------------------------------------------------
typedef __attribute__((ext_vector_type(16))) __bf16 bf16x16;
typedef __attribute__((ext_vector_type(8)))  float  v8f;
typedef __attribute__((ext_vector_type(4)))  int    i32x4;

union FragBf {
    uint4    q[2];
    bf16x16  v;
};

__device__ __forceinline__ unsigned short f2bf(float f) {
    union { float f; unsigned u; } x;
    x.f = f;
    unsigned u = x.u;
    unsigned r = (u + 0x7FFFu + ((u >> 16) & 1u)) >> 16;   // round-nearest-even
    return (unsigned short)r;
}

// Packed f32x2 -> bf16x2 (single v_cvt_pk_bf16_f32 when available)
__device__ __forceinline__ unsigned pk2bf(float a, float b) {
#if __has_builtin(__builtin_amdgcn_cvt_pk_bf16_f32)
    auto p = __builtin_amdgcn_cvt_pk_bf16_f32(a, b);
    return __builtin_bit_cast(unsigned, p);
#else
    return (unsigned)f2bf(a) | ((unsigned)f2bf(b) << 16);
#endif
}

// 16-byte global -> LDS copy, async (ASYNCcnt) when the builtin exists.
#if __has_builtin(__builtin_amdgcn_global_load_async_to_lds_b128) && \
    __has_builtin(__builtin_amdgcn_s_wait_asynccnt)
#define HAS_ASYNC_LDS 1
#else
#define HAS_ASYNC_LDS 0
#endif

__device__ __forceinline__ void cp16_g2l(void* lds, const void* g) {
#if HAS_ASYNC_LDS
    __builtin_amdgcn_global_load_async_to_lds_b128(
        (__attribute__((address_space(1))) i32x4*)g,
        (__attribute__((address_space(3))) i32x4*)lds, 0, 0);
#else
    *reinterpret_cast<uint4*>(lds) = *reinterpret_cast<const uint4*>(g);
#endif
}
__device__ __forceinline__ void async_wait() {
#if HAS_ASYNC_LDS
    __builtin_amdgcn_s_wait_asynccnt(0);
#endif
}

__device__ __forceinline__ v8f wmma_bf16(const bf16x16& a, const bf16x16& b, const v8f& c) {
    return __builtin_amdgcn_wmma_f32_16x16x32_bf16(false, a, false, b, (short)0, c, false, false);
}

// ---------------------------------------------------------------------------
// Weight pre-pass: wt[n*K + k] = bf16(w[k*N + n])  (transpose into the exact
// [n][k] layout the WMMA B-fragment reads, so B tiles are contiguous rows
// and can be staged with async b128 copies)
// ---------------------------------------------------------------------------
__global__ __launch_bounds__(256) void transpose_w(const float* __restrict__ w,
                                                   unsigned short* __restrict__ wt,
                                                   int K, int N) {
    size_t idx = (size_t)blockIdx.x * 256 + threadIdx.x;
    if (idx >= (size_t)N * K) return;
    int n = (int)(idx / K);
    int k = (int)(idx % K);
    wt[idx] = f2bf(w[(size_t)k * N + n]);
}

// ---------------------------------------------------------------------------
// bf16 WMMA GEMM: C[M,N] = A[M,K] * BT^T + bias (+ReLU).  BT is bf16 [N][K].
// A is f32 (converted while staging) or bf16 (async-staged) per ABF16.
// Output f32 or bf16 per BF16OUT.  Block 256 thr = 8 waves; tile 128x128x32.
// ---------------------------------------------------------------------------
template <bool RELU, bool BF16OUT, bool ABF16>
__global__ __launch_bounds__(256) void gemm_bf16(const void* __restrict__ Ain,
                                                 const unsigned short* __restrict__ BT,
                                                 const float* __restrict__ bias,
                                                 void* __restrict__ Cout,
                                                 int M, int N, int K) {
    __shared__ __attribute__((aligned(16))) unsigned short As[128 * 32];
    __shared__ __attribute__((aligned(16))) unsigned short Bs[128 * 32];   // [n][k]

    const int tid  = threadIdx.x;
    const int nb   = blockIdx.x * 128;
    const int mb   = blockIdx.y * 128;
    const int wid  = tid >> 5;
    const int lane = tid & 31;
    const int hs   = lane >> 4;
    const int ln   = lane & 15;
    const int wm   = (wid >> 2) * 64;
    const int wn   = (wid & 3)  * 32;

    const float*          Af = (const float*)Ain;
    const unsigned short* Ab = (const unsigned short*)Ain;

    v8f zero8 = {0.f, 0.f, 0.f, 0.f, 0.f, 0.f, 0.f, 0.f};
    v8f acc[4][2];
#pragma unroll
    for (int mf = 0; mf < 4; ++mf)
#pragma unroll
        for (int nf = 0; nf < 2; ++nf) acc[mf][nf] = zero8;

    for (int kb = 0; kb < K; kb += 32) {
        // ---- stage A tile (128 rows x 32 k)
        if constexpr (ABF16) {
#pragma unroll
            for (int i = 0; i < 2; ++i) {
                int idx  = tid + i * 256;            // 512 x 16B
                int r    = idx >> 2;
                int part = idx & 3;
                cp16_g2l(&As[r * 32 + part * 8],
                         Ab + (size_t)(mb + r) * K + kb + part * 8);
            }
        } else {
#pragma unroll
            for (int i = 0; i < 4; ++i) {
                int idx = tid + i * 256;
                int r   = idx >> 3;
                int c   = (idx & 7) * 4;
                float4 av = *reinterpret_cast<const float4*>(Af + (size_t)(mb + r) * K + kb + c);
                uint2 p;
                p.x = pk2bf(av.x, av.y);
                p.y = pk2bf(av.z, av.w);
                *reinterpret_cast<uint2*>(&As[r * 32 + c]) = p;
            }
        }
        // ---- stage B tile: BT rows are already [n][k] -> straight async copy
#pragma unroll
        for (int i = 0; i < 2; ++i) {
            int idx  = tid + i * 256;                // 512 x 16B
            int r    = idx >> 2;
            int part = idx & 3;
            cp16_g2l(&Bs[r * 32 + part * 8],
                     BT + (size_t)(nb + r) * K + kb + part * 8);
        }
        if (kb + 32 < K) {   // prefetch next tiles (global_prefetch_b8)
            __builtin_prefetch(BT + (size_t)(nb + (tid & 127)) * K + kb + 32, 0, 1);
            if (!ABF16)
                __builtin_prefetch(Af + (size_t)(mb + (tid & 127)) * K + kb + 32, 0, 1);
        }
        async_wait();
        __syncthreads();

        FragBf bfr[2];
#pragma unroll
        for (int nf = 0; nf < 2; ++nf) {
            const unsigned short* p = &Bs[(wn + nf * 16 + ln) * 32 + hs * 16];
            bfr[nf].q[0] = *reinterpret_cast<const uint4*>(p);
            bfr[nf].q[1] = *reinterpret_cast<const uint4*>(p + 8);
        }
#pragma unroll
        for (int mf = 0; mf < 4; ++mf) {
            FragBf af;
            const unsigned short* p = &As[(wm + mf * 16 + ln) * 32];
            af.q[0] = *reinterpret_cast<const uint4*>(p + hs * 16);
            af.q[1] = *reinterpret_cast<const uint4*>(p + hs * 16 + 8);
#pragma unroll
            for (int nf = 0; nf < 2; ++nf)
                acc[mf][nf] = wmma_bf16(af.v, bfr[nf].v, acc[mf][nf]);
        }
        __syncthreads();
    }

#pragma unroll
    for (int mf = 0; mf < 4; ++mf)
#pragma unroll
        for (int nf = 0; nf < 2; ++nf) {
            int n = nb + wn + nf * 16 + ln;
            float bv = bias[n];
#pragma unroll
            for (int r = 0; r < 8; ++r) {
                int m = mb + wm + mf * 16 + hs * 8 + r;
                float val = acc[mf][nf][r] + bv;
                if (RELU) val = fmaxf(val, 0.f);
                if constexpr (BF16OUT)
                    ((unsigned short*)Cout)[(size_t)m * N + n] = f2bf(val);
                else
                    ((float*)Cout)[(size_t)m * N + n] = val;
            }
        }
}

// ---------------------------------------------------------------------------
// Flash attention over bf16 qkv. Grid (N/128, B*H), 256 thr = 8 waves.
// qkv row layout per token (bf16, stride 3072): [head][q64 k64 v64]
// ---------------------------------------------------------------------------
__global__ __launch_bounds__(256) void attn_kernel(const unsigned short* __restrict__ qkv,
                                                   float* __restrict__ out) {
    extern __shared__ unsigned short smem[];
    unsigned short* Qs = smem;                    // 128 x 64
    unsigned short* Ks = smem + 128 * 64;         // 128 x 64
    unsigned short* Vt = smem + 2 * 128 * 64;     // 64 x 128 (transposed V)
    unsigned short* Ps = smem + 3 * 128 * 64;     // 8 x 16 x 128

    const int tid  = threadIdx.x;
    const int wid  = tid >> 5;
    const int lane = tid & 31;
    const int hs   = lane >> 4;
    const int ln   = lane & 15;
    const int qb   = blockIdx.x * 128;
    const int bh   = blockIdx.y;
    const int b    = bh >> 4;
    const int h    = bh & 15;
    const float sscale = 0.125f;                  // 1/sqrt(64), applied in softmax
    const size_t headOff = (size_t)h * 192;

    // ---- stage Q via async b128 copies (1024 x 16B)
#pragma unroll
    for (int i = 0; i < 4; ++i) {
        int idx  = tid + i * 256;
        int r    = idx >> 3;
        int part = idx & 7;
        cp16_g2l(&Qs[r * 64 + part * 8],
                 qkv + (size_t)(b * 2048 + qb + r) * 3072 + headOff + part * 8);
    }

    float mrow[8], lrow[8];
    v8f zero8 = {0.f, 0.f, 0.f, 0.f, 0.f, 0.f, 0.f, 0.f};
    v8f oacc[4];
#pragma unroll
    for (int r = 0; r < 8; ++r) { mrow[r] = -3.0e38f; lrow[r] = 0.f; }
#pragma unroll
    for (int nf = 0; nf < 4; ++nf) oacc[nf] = zero8;
    async_wait();
    __syncthreads();

    unsigned short* Pw = Ps + (size_t)wid * 16 * 128;

    for (int kb = 0; kb < 16; ++kb) {
        // ---- stage K rows (async) and V transposed (vector load + scatter)
#pragma unroll
        for (int i = 0; i < 4; ++i) {
            int idx  = tid + i * 256;
            int r    = idx >> 3;
            int part = idx & 7;
            size_t base = (size_t)(b * 2048 + kb * 128 + r) * 3072 + headOff;
            cp16_g2l(&Ks[r * 64 + part * 8], qkv + base + 64 + part * 8);
            int c = part * 8;
            union { uint4 q; unsigned short s[8]; } t;
            t.q = *reinterpret_cast<const uint4*>(qkv + base + 128 + c);
#pragma unroll
            for (int j = 0; j < 8; ++j) Vt[(c + j) * 128 + r] = t.s[j];
        }
        if (kb + 1 < 16)
            __builtin_prefetch(qkv + (size_t)(b * 2048 + (kb + 1) * 128 + (tid & 127)) * 3072
                                   + headOff + 64, 0, 1);
        async_wait();
        __syncthreads();

        // ---- S = Q K^T : 16 rows x 128 keys per wave
        v8f sacc[8];
#pragma unroll
        for (int f = 0; f < 8; ++f) sacc[f] = zero8;
#pragma unroll
        for (int kk = 0; kk < 64; kk += 32) {
            FragBf af;
            const unsigned short* ap = &Qs[(wid * 16 + ln) * 64 + kk];
            af.q[0] = *reinterpret_cast<const uint4*>(ap + hs * 16);
            af.q[1] = *reinterpret_cast<const uint4*>(ap + hs * 16 + 8);
#pragma unroll
            for (int f = 0; f < 8; ++f) {
                FragBf bf_;
                const unsigned short* bp = &Ks[(f * 16 + ln) * 64 + kk + hs * 16];
                bf_.q[0] = *reinterpret_cast<const uint4*>(bp);
                bf_.q[1] = *reinterpret_cast<const uint4*>(bp + 8);
                sacc[f] = wmma_bf16(af.v, bf_.v, sacc[f]);
            }
        }

        // ---- online softmax (scale folded into exp; max commutes with scale)
#pragma unroll
        for (int r = 0; r < 8; ++r) {
            float mx = -3.0e38f;
#pragma unroll
            for (int f = 0; f < 8; ++f) mx = fmaxf(mx, sacc[f][r]);
            mx = fmaxf(mx, __shfl_xor(mx, 1, 32));
            mx = fmaxf(mx, __shfl_xor(mx, 2, 32));
            mx = fmaxf(mx, __shfl_xor(mx, 4, 32));
            mx = fmaxf(mx, __shfl_xor(mx, 8, 32));
            float nm    = fmaxf(mrow[r], mx * sscale);
            float alpha = __expf(mrow[r] - nm);
            float rs = 0.f;
#pragma unroll
            for (int f = 0; f < 8; ++f) {
                float p = __expf(fmaf(sacc[f][r], sscale, -nm));
                sacc[f][r] = p;
                rs += p;
            }
            rs += __shfl_xor(rs, 1, 32);
            rs += __shfl_xor(rs, 2, 32);
            rs += __shfl_xor(rs, 4, 32);
            rs += __shfl_xor(rs, 8, 32);
            lrow[r] = lrow[r] * alpha + rs;
            mrow[r] = nm;
#pragma unroll
            for (int nf = 0; nf < 4; ++nf) oacc[nf][r] *= alpha;
        }

        // ---- P -> LDS (wave-private), re-layout C -> A
#pragma unroll
        for (int f = 0; f < 8; ++f)
#pragma unroll
            for (int r = 0; r < 8; ++r)
                Pw[(hs * 8 + r) * 128 + f * 16 + ln] = f2bf(sacc[f][r]);

        // ---- O += P * V
#pragma unroll
        for (int kk4 = 0; kk4 < 4; ++kk4) {
            FragBf af;
            const unsigned short* ap = Pw + ln * 128 + kk4 * 32;
            af.q[0] = *reinterpret_cast<const uint4*>(ap + hs * 16);
            af.q[1] = *reinterpret_cast<const uint4*>(ap + hs * 16 + 8);
#pragma unroll
            for (int nf = 0; nf < 4; ++nf) {
                FragBf bf_;
                const unsigned short* bp = &Vt[(nf * 16 + ln) * 128 + kk4 * 32 + hs * 16];
                bf_.q[0] = *reinterpret_cast<const uint4*>(bp);
                bf_.q[1] = *reinterpret_cast<const uint4*>(bp + 8);
                oacc[nf] = wmma_bf16(af.v, bf_.v, oacc[nf]);
            }
        }
        __syncthreads();
    }

#pragma unroll
    for (int nf = 0; nf < 4; ++nf)
#pragma unroll
        for (int r = 0; r < 8; ++r) {
            int row = qb + wid * 16 + hs * 8 + r;
            int d   = nf * 16 + ln;
            out[(size_t)(b * 2048 + row) * 1024 + h * 64 + d] = oacc[nf][r] / lrow[r];
        }
}

// ---------------------------------------------------------------------------
// LayerNorm with fused residual: out = LN(src + res) * g + beta   (row = 1024)
// ---------------------------------------------------------------------------
__global__ __launch_bounds__(256) void ln_kernel(const float* __restrict__ src,
                                                 const float* __restrict__ res,
                                                 const float* __restrict__ g,
                                                 const float* __restrict__ be,
                                                 float* __restrict__ out) {
    __shared__ float red[256];
    const int row = blockIdx.x;
    const int tid = threadIdx.x;
    const size_t base = (size_t)row * 1024;

    float v[4];
    float s = 0.f;
#pragma unroll
    for (int i = 0; i < 4; ++i) {
        int idx = tid + i * 256;
        float t = src[base + idx] + res[base + idx];
        v[i] = t;
        s += t;
    }
    red[tid] = s;
    __syncthreads();
    for (int o = 128; o > 0; o >>= 1) {
        if (tid < o) red[tid] += red[tid + o];
        __syncthreads();
    }
    float mu = red[0] * (1.f / 1024.f);
    __syncthreads();

    float s2 = 0.f;
#pragma unroll
    for (int i = 0; i < 4; ++i) {
        float d = v[i] - mu;
        s2 += d * d;
    }
    red[tid] = s2;
    __syncthreads();
    for (int o = 128; o > 0; o >>= 1) {
        if (tid < o) red[tid] += red[tid + o];
        __syncthreads();
    }
    float rstd = rsqrtf(red[0] * (1.f / 1024.f) + 1e-5f);

#pragma unroll
    for (int i = 0; i < 4; ++i) {
        int idx = tid + i * 256;
        out[base + idx] = (v[i] - mu) * rstd * g[idx] + be[idx];
    }
}

// ---------------------------------------------------------------------------
// Launcher
// ---------------------------------------------------------------------------
extern "C" void kernel_launch(void* const* d_in, const int* in_sizes, int n_in,
                              void* d_out, int out_size, void* d_ws, size_t ws_size,
                              hipStream_t stream) {
    const float* x      = (const float*)d_in[0];
    const float* w_qkv  = (const float*)d_in[1];
    const float* b_qkv  = (const float*)d_in[2];
    const float* w_proj = (const float*)d_in[3];
    const float* b_proj = (const float*)d_in[4];
    const float* g1     = (const float*)d_in[5];
    const float* be1    = (const float*)d_in[6];
    const float* w_ff1  = (const float*)d_in[7];
    const float* b_ff1  = (const float*)d_in[8];
    const float* w_ff2  = (const float*)d_in[9];
    const float* b_ff2  = (const float*)d_in[10];
    const float* g2     = (const float*)d_in[11];
    const float* be2    = (const float*)d_in[12];
    float* out = (float*)d_out;

    // ---- workspace carve-up (bytes)
    char* p = (char*)d_ws;
    unsigned short* qkvB   = (unsigned short*)p; p += (size_t)4096 * 3072 * 2;  // bf16
    unsigned short* wqkvT  = (unsigned short*)p; p += (size_t)3072 * 1024 * 2;
    unsigned short* wprojT = (unsigned short*)p; p += (size_t)1024 * 1024 * 2;
    unsigned short* wff1T  = (unsigned short*)p; p += (size_t)4096 * 1024 * 2;
    unsigned short* wff2T  = (unsigned short*)p; p += (size_t)1024 * 4096 * 2;
    float*          tmp    = (float*)p;          p += (size_t)4096 * 1024 * 4;  // f32
    float*          hbuf   = (float*)p;          p += (size_t)4096 * 1024 * 4;
    unsigned short* ff1B   = (unsigned short*)p; p += (size_t)4096 * 4096 * 2;  // bf16

    dim3 blk(256);

    // 0. weight pre-pass: f32 [K][N] -> bf16 [N][K]
    transpose_w<<<dim3((3072 * 1024) / 256), blk, 0, stream>>>(w_qkv,  wqkvT, 1024, 3072);
    transpose_w<<<dim3((1024 * 1024) / 256), blk, 0, stream>>>(w_proj, wprojT, 1024, 1024);
    transpose_w<<<dim3((4096 * 1024) / 256), blk, 0, stream>>>(w_ff1,  wff1T, 1024, 4096);
    transpose_w<<<dim3((4096 * 1024) / 256), blk, 0, stream>>>(w_ff2,  wff2T, 4096, 1024);

    // 1. QKV projection -> bf16 qkv
    gemm_bf16<false, true, false><<<dim3(24, 32), blk, 0, stream>>>(
        x, wqkvT, b_qkv, qkvB, 4096, 3072, 1024);

    // 2. Attention (80 KB dynamic LDS)
    attn_kernel<<<dim3(16, 32), blk, 80 * 1024, stream>>>(qkvB, tmp);

    // 3. Output projection -> f32 hbuf
    gemm_bf16<false, false, false><<<dim3(8, 32), blk, 0, stream>>>(
        tmp, wprojT, b_proj, hbuf, 4096, 1024, 1024);

    // 4. h = LN(proj + x) in place
    ln_kernel<<<dim3(4096), blk, 0, stream>>>(hbuf, x, g1, be1, hbuf);

    // 5. FF1 + ReLU -> bf16
    gemm_bf16<true, true, false><<<dim3(32, 32), blk, 0, stream>>>(
        hbuf, wff1T, b_ff1, ff1B, 4096, 4096, 1024);

    // 6. FF2 (bf16 A, async-staged) -> f32 tmp
    gemm_bf16<false, false, true><<<dim3(8, 32), blk, 0, stream>>>(
        ff1B, wff2T, b_ff2, tmp, 4096, 1024, 4096);

    // 7. out = LN(ff + h)
    ln_kernel<<<dim3(4096), blk, 0, stream>>>(tmp, hbuf, g2, be2, out);
}
